// MultiScalePeriodicAttentionLayer_40037685133873
// MI455X (gfx1250) — matrix operand order repaired and checked by reading, not compiled, not run on hardware
//
#include <hip/hip_runtime.h>
#include <math.h>
#include <stdint.h>

typedef __attribute__((ext_vector_type(16))) __bf16 v16bf;
typedef __attribute__((ext_vector_type(8)))  float  v8f;

#define BATCH   8
#define SEQLEN  16
#define HW      1024   // 32*32
#define DMODEL  128
#define NEXP    6
#define NHEADS  8
#define HDIM    16

__device__ __forceinline__ __bf16 f2bf(float f) {
    return (__bf16)f;    // native v_cvt bf16 (RNE)
}
__device__ __forceinline__ float bf2f(__bf16 b) {
    unsigned short h = __builtin_bit_cast(unsigned short, b);
    unsigned u = ((unsigned)h) << 16;
    return __builtin_bit_cast(float, u);
}

// ---------------------------------------------------------------- gate path
__global__ void pool_kernel(const float* __restrict__ x, float* __restrict__ pooled) {
    int bt = blockIdx.x;      // 0..127  (b*16+t)
    int d  = threadIdx.x;     // 0..127
    const float* base = x + ((size_t)bt * HW) * DMODEL + d;
    float s = 0.f;
    for (int hw = 0; hw < HW; ++hw) s += base[(size_t)hw * DMODEL];
    pooled[bt * DMODEL + d] = s * (1.0f / HW);
}

__global__ void gate_kernel(const float* __restrict__ pooled,
                            const float* __restrict__ wg,
                            float* __restrict__ gates) {
    __shared__ float amp[BATCH][8];
    int tid = threadIdx.x;    // 64 threads
    if (tid < 64) {
        int b = tid >> 3, k = (tid & 7) + 1;
        float cs[SEQLEN], sn[SEQLEN];
        for (int t = 0; t < SEQLEN; ++t) {
            float ang = -6.283185307179586f * (float)(k * t) * (1.0f / 16.0f);
            __sincosf(ang, &sn[t], &cs[t]);
        }
        const float* pb = pooled + b * SEQLEN * DMODEL;
        float a = 0.f;
        for (int d = 0; d < DMODEL; ++d) {
            float re = 0.f, im = 0.f;
            for (int t = 0; t < SEQLEN; ++t) {
                float v = pb[t * DMODEL + d];
                re += v * cs[t]; im += v * sn[t];
            }
            a += sqrtf(re * re + im * im);
        }
        amp[b][k - 1] = a * (0.25f / (float)DMODEL);   // ortho norm 1/sqrt(16)
    }
    __syncthreads();
    if (tid < BATCH) {
        int b = tid;
        float logits[NEXP];
        for (int e = 0; e < NEXP; ++e) {
            float s = 0.f;
            for (int k = 0; k < 8; ++k) s += amp[b][k] * wg[k * NEXP + e];
            logits[e] = s;
        }
        int i0 = 0;
        for (int e = 1; e < NEXP; ++e) if (logits[e] > logits[i0]) i0 = e;
        int i1 = -1;
        for (int e = 0; e < NEXP; ++e) {
            if (e == i0) continue;
            if (i1 < 0 || logits[e] > logits[i1]) i1 = e;
        }
        float m = logits[i0];
        float e0 = __expf(logits[i0] - m), e1 = __expf(logits[i1] - m);
        float inv = 1.f / (e0 + e1);
        for (int e = 0; e < NEXP; ++e) gates[b * NEXP + e] = 0.f;
        gates[b * NEXP + i0] = e0 * inv;
        gates[b * NEXP + i1] = e1 * inv;
    }
}

// ------------------------------------------------------- weight prep (bf16, transposed)
// WT[c*R + r] = bf16(W[r*C + c])
__global__ void transpose_kernel(const float* __restrict__ W, __bf16* __restrict__ WT,
                                 int R, int C) {
    int idx = blockIdx.x * blockDim.x + threadIdx.x;
    if (idx >= R * C) return;
    int c = idx / R, r = idx - c * R;
    WT[(size_t)c * R + r] = f2bf(W[(size_t)r * C + c]);
}

__global__ void zero_kernel(float* __restrict__ p, int nelem) {
    int i = blockIdx.x * blockDim.x + threadIdx.x;
    if (i < nelem) p[i] = 0.f;
}

__global__ void log_kernel(float* __restrict__ p, int nelem) {
    int i = blockIdx.x * blockDim.x + threadIdx.x;
    if (i < nelem) {
        float v = p[i];
        if (v == 0.f) v = 2.220446049250313e-16f;
        p[i] = __logf(v);
    }
}

// ---------------------------------------------------------------- expert kernel
// One block = 16 rows r=(b,hw). 8 waves of 32 lanes. Dynamic LDS layout:
//   segf : 16 x Tp x 128 fp32  (async-staged; region reused as O bf16 later)
//   segbf: 16 x Tp x 128 bf16  (converted once, feeds WMMA A fragments)
//   q/k/v: 16 x n  x 128 bf16 each
__global__ __launch_bounds__(256) void expert_kernel(
        const float* __restrict__ x,
        const __bf16* __restrict__ WqT, const __bf16* __restrict__ WkT,
        const __bf16* __restrict__ WvT, const __bf16* __restrict__ WoT,
        const float* __restrict__ gates, float* __restrict__ out,
        int e, int p, int n) {
    extern __shared__ char smem[];
    const int Tp = n * p;          // >= 16 for every expert
    const int K  = p * DMODEL;
    const int segStride = Tp * DMODEL;   // elements per seg row

    float*  segf  = (float*)smem;
    __bf16* segbf = (__bf16*)(smem + (size_t)16 * segStride * 4);
    __bf16* qs    = segbf + 16 * segStride;
    __bf16* ks    = qs + 16 * n * DMODEL;
    __bf16* vs    = ks + 16 * n * DMODEL;
    __bf16* obf   = (__bf16*)smem;   // alias of segf (dead after stage 1b)

    const int tid  = threadIdx.x;
    const int lane = tid & 31;
    const int wv   = tid >> 5;
    const int r0   = blockIdx.x * 16;
    const int b    = r0 / HW;
    const int hw0  = r0 % HW;
    const float gate = gates[b * NEXP + e];

    // ---- stage 1: async-stage seg tile (t < 16) straight into LDS, no VGPR round-trip
    {
        // 16 rows x 16 t x 128 floats = 16 rows x 16 t x 32 chunks of 16B
        const int chunks = 16 * SEQLEN * 32;     // 8192
        for (int cidx = tid; cidx < chunks; cidx += 256) {
            int row = cidx >> 9;          // / 512
            int rem = cidx & 511;
            int t   = rem >> 5;
            int ch  = rem & 31;
            const float* gsrc =
                x + (((size_t)(b * SEQLEN + t)) * HW + (hw0 + row)) * DMODEL + ch * 4;
            float* ldst = segf + row * segStride + t * DMODEL + ch * 4;
            unsigned loff = (unsigned)(uintptr_t)ldst;            // LDS byte offset
            unsigned long long gaddr = (unsigned long long)(uintptr_t)gsrc;
            asm volatile("global_load_async_to_lds_b128 %0, %1, off"
                         :: "v"(loff), "v"(gaddr) : "memory");
        }
        // zero-fill the padded tail rows (t in [16, Tp)) with plain DS stores
        const int padElems = 16 * (Tp - SEQLEN) * DMODEL;
        for (int idx = tid; idx < padElems; idx += 256) {
            int rowLen = (Tp - SEQLEN) * DMODEL;
            int row = idx / rowLen;
            int rem = idx - row * rowLen;
            segf[row * segStride + SEQLEN * DMODEL + rem] = 0.f;
        }
        asm volatile("s_wait_asynccnt 0x0" ::: "memory");
    }
    __syncthreads();

    // ---- stage 1b: one-shot fp32 -> bf16 conversion of the seg tile (packed pairs)
    {
        const int totalPairs = 16 * segStride / 2;
        unsigned* dst = (unsigned*)segbf;
        for (int idx = tid; idx < totalPairs; idx += 256) {
            float lo = segf[2 * idx], hi = segf[2 * idx + 1];
            unsigned short l = __builtin_bit_cast(unsigned short, f2bf(lo));
            unsigned short h = __builtin_bit_cast(unsigned short, f2bf(hi));
            dst[idx] = (unsigned)l | ((unsigned)h << 16);
        }
    }
    __syncthreads();

    // ---- stage 2: Q/K/V = seg @ W via WMMA, rotary on q,k
    {
        const int col  = 16 * wv + (lane & 15);
        const int hiA  = (lane >= 16) ? 8 : 0;
        const int hiB  = (lane >= 16) ? 8 : 0;
        const int rowA = lane & 15;
        const int j    = col >> 1;
        const float freq = __expf(-(float)j * 0.14391156064519868f); // ln(1e4)/64
        const float sgn  = (col & 1) ? 1.0f : -1.0f;

        for (int i = 0; i < n; ++i) {
            v8f accq = {}, acck = {}, accv = {};
            const __bf16* segRow = segbf + rowA * segStride + i * K;
            for (int kk = 0; kk < K; kk += 32) {
                v16bf a, bq, bk, bv;
#pragma unroll
                for (int jj = 0; jj < 8; ++jj) {
                    a[jj]     = segRow[kk + hiA + jj];
                    a[jj + 8] = segRow[kk + hiA + 16 + jj];
                }
                const __bf16* wq = WqT + (size_t)col * K + kk + hiB;
                const __bf16* wk = WkT + (size_t)col * K + kk + hiB;
                const __bf16* wp = WvT + (size_t)col * K + kk + hiB;
                __builtin_prefetch(wq + 512, 0, 3);
#pragma unroll
                for (int jj = 0; jj < 8; ++jj) {
                    bq[jj] = wq[jj]; bk[jj] = wk[jj]; bv[jj] = wp[jj];
                    bq[jj + 8] = wq[16 + jj]; bk[jj + 8] = wk[16 + jj]; bv[jj + 8] = wp[16 + jj];
                }
                accq = __builtin_amdgcn_wmma_f32_16x16x32_bf16(false, a, false, bq, (short)0, accq, false, false);
                acck = __builtin_amdgcn_wmma_f32_16x16x32_bf16(false, a, false, bk, (short)0, acck, false, false);
                accv = __builtin_amdgcn_wmma_f32_16x16x32_bf16(false, a, false, bv, (short)0, accv, false, false);
            }
            float s, c;
            __sincosf((float)i * freq, &s, &c);
#pragma unroll
            for (int g = 0; g < 8; ++g) {
                int rL = g + hiA;
                float vq = accq[g];
                float pq = __shfl_xor(vq, 1, 32);
                float nq = vq * c + sgn * pq * s;
                float vk = acck[g];
                float pk = __shfl_xor(vk, 1, 32);
                float nk = vk * c + sgn * pk * s;
                int off = (rL * n + i) * DMODEL + col;
                qs[off] = f2bf(nq);
                ks[off] = f2bf(nk);
                vs[off] = f2bf(accv[g]);
            }
        }
    }
    __syncthreads();

    // ---- stage 3: tiny n x n attention per (row, head); O -> obf (bf16, recycled seg)
    if (tid < 128) {
        const int rL = tid >> 3;
        const int h  = tid & 7;
        const int d0 = h * HDIM;
        const __bf16* qr = qs + rL * n * DMODEL;
        const __bf16* kr = ks + rL * n * DMODEL;
        const __bf16* vr = vs + rL * n * DMODEL;
        for (int ii = 0; ii < n; ++ii) {
            float sc[8];
            float mx = -1e30f;
            for (int jj = 0; jj < n; ++jj) {
                float ssum = 0.f;
                for (int d = 0; d < HDIM; ++d)
                    ssum += bf2f(qr[ii * DMODEL + d0 + d]) * bf2f(kr[jj * DMODEL + d0 + d]);
                ssum *= 0.25f;   // 1/sqrt(HDIM)
                sc[jj] = ssum;
                mx = fmaxf(mx, ssum);
            }
            float denom = 0.f;
            for (int jj = 0; jj < n; ++jj) { sc[jj] = __expf(sc[jj] - mx); denom += sc[jj]; }
            float inv = 1.f / denom;
            for (int d = 0; d < HDIM; ++d) {
                float o = 0.f;
                for (int jj = 0; jj < n; ++jj)
                    o += sc[jj] * bf2f(vr[jj * DMODEL + d0 + d]);
                obf[(rL * n + ii) * DMODEL + d0 + d] = f2bf(o * inv);
            }
        }
    }
    __syncthreads();

    // ---- stage 4: O @ Wo via WMMA, fused gate*exp accumulation into out
    {
        const int hiA  = (lane >= 16) ? 8 : 0;
        const int hiB  = (lane >= 16) ? 8 : 0;
        const int rowA = lane & 15;
        const int ntiles = K / 16;   // p*8 N-tiles
        for (int jt = wv; jt < ntiles; jt += 8) {
            const int col  = jt * 16 + (lane & 15);
            const int tsub = col >> 7;
            const int dcol = col & 127;
            const __bf16* wo = WoT + (size_t)col * DMODEL;
            for (int i = 0; i < n; ++i) {
                v8f acc = {};
                const __bf16* orow = obf + (rowA * n + i) * DMODEL;
                for (int kk = 0; kk < DMODEL; kk += 32) {
                    v16bf a, bo;
#pragma unroll
                    for (int jj = 0; jj < 8; ++jj) {
                        a[jj]     = orow[kk + hiA + jj];
                        a[jj + 8] = orow[kk + hiA + 16 + jj];
                    }
#pragma unroll
                    for (int jj = 0; jj < 8; ++jj) { bo[jj] = wo[kk + hiB + jj]; bo[jj + 8] = wo[kk + hiB + 16 + jj]; }
                    acc = __builtin_amdgcn_wmma_f32_16x16x32_bf16(false, a, false, bo, (short)0, acc, false, false);
                }
                const int t = i * p + tsub;
                if (t < SEQLEN) {
#pragma unroll
                    for (int g = 0; g < 8; ++g) {
                        int rL = g + hiA;
                        size_t oidx = (((size_t)(b * SEQLEN + t)) * HW + (hw0 + rL)) * DMODEL + dcol;
                        out[oidx] += gate * __expf(acc[g]);
                    }
                }
            }
        }
    }
}

// ---------------------------------------------------------------- launch
static const int SEG_P[NEXP] = {16, 8, 5, 4, 3, 2};
static const int SEG_N[NEXP] = {1, 2, 4, 4, 6, 8};

extern "C" void kernel_launch(void* const* d_in, const int* in_sizes, int n_in,
                              void* d_out, int out_size, void* d_ws, size_t ws_size,
                              hipStream_t stream) {
    (void)in_sizes; (void)n_in; (void)ws_size;
    const float* x  = (const float*)d_in[0];
    const float* wg = (const float*)d_in[1];
    float* out = (float*)d_out;
    char*  ws  = (char*)d_ws;

    float* gates  = (float*)ws;                       // 48 floats
    float* pooled = (float*)(ws + 256);               // B*T*D floats
    char*  wts    = ws + 256 + (size_t)BATCH * SEQLEN * DMODEL * 4;

    // gate path
    pool_kernel<<<BATCH * SEQLEN, DMODEL, 0, stream>>>(x, pooled);
    gate_kernel<<<1, 64, 0, stream>>>(pooled, wg, gates);

    // transposed bf16 weights (recomputed every call: deterministic)
    __bf16* WT[NEXP][4];
    size_t off = 0;
    for (int e = 0; e < NEXP; ++e) {
        int K = SEG_P[e] * DMODEL;
        for (int m = 0; m < 4; ++m) {
            const float* W = (const float*)d_in[2 + 4 * e + m];
            WT[e][m] = (__bf16*)(wts + off);
            off += (size_t)K * DMODEL * 2;
            off = (off + 255) & ~(size_t)255;
            int R = (m < 3) ? K : DMODEL;
            int C = (m < 3) ? DMODEL : K;
            transpose_kernel<<<(R * C + 255) / 256, 256, 0, stream>>>(W, WT[e][m], R, C);
        }
    }

    zero_kernel<<<(out_size + 255) / 256, 256, 0, stream>>>(out, out_size);

    for (int e = 0; e < NEXP; ++e) {
        int p = SEG_P[e], n = SEG_N[e];
        int Tp = p * n;
        size_t shmem = (size_t)16 * Tp * DMODEL * 4     // segf fp32 (later O bf16 alias)
                     + (size_t)16 * Tp * DMODEL * 2     // segbf bf16
                     + (size_t)3 * 16 * n * DMODEL * 2; // q,k,v bf16
        expert_kernel<<<8 * HW / 16, 256, shmem, stream>>>(
            x, WT[e][0], WT[e][1], WT[e][2], WT[e][3], gates, out, e, p, n);
    }

    log_kernel<<<(out_size + 255) / 256, 256, 0, stream>>>(out, out_size);
}
